// MultiHeadAttention_60911226191904
// MI455X (gfx1250) — matrix operand order repaired and checked by reading, not compiled, not run on hardware
//
#include <hip/hip_runtime.h>
#include <hip/hip_bf16.h>

// ---------- types ----------
typedef __attribute__((ext_vector_type(16))) __bf16 v16bf;
typedef __attribute__((ext_vector_type(8)))  float  v8f;

union Frag {
    unsigned int u[8];
    v16bf v;
};

static __device__ __forceinline__ unsigned short f2bf(float f) {
    unsigned int u = __float_as_uint(f);
    unsigned int r = (u + 0x7FFFu + ((u >> 16) & 1u)) >> 16;   // RNE
    return (unsigned short)r;
}

static __device__ __forceinline__ v8f vzero8() {
    v8f z = {0.f, 0.f, 0.f, 0.f, 0.f, 0.f, 0.f, 0.f};
    return z;
}

// ---------- CDNA5 async memory->LDS (ASYNCcnt) ----------
#define USE_ASYNC 1

static __device__ __forceinline__ void async_copy16(void* lds_dst, const void* gsrc) {
#if USE_ASYNC
    // generic LDS pointer: low 32 bits are the DS-usable address
    unsigned int lo = (unsigned int)(unsigned long long)lds_dst;
    asm volatile("global_load_async_to_lds_b128 %0, %1, off"
                 :: "v"(lo), "v"(gsrc) : "memory");
#else
    const unsigned int* s = (const unsigned int*)gsrc;
    unsigned int* d = (unsigned int*)lds_dst;
#pragma unroll
    for (int i = 0; i < 4; ++i) d[i] = s[i];
#endif
}

static __device__ __forceinline__ void wait_async0() {
#if USE_ASYNC
    asm volatile("s_wait_asynccnt 0x0" ::: "memory");
#endif
}

// ---------- sizes ----------
#define BATCH   16
#define SEQ     1024
#define EMBED   384
#define HEADS   6
#define HDIM    64
#define ROWS    (BATCH * SEQ)          // 16384
#define QKV_N   (3 * EMBED)            // 1152
#define ATTN_SCALE 0.125f              // 1/sqrt(64)

// =====================================================================
// fp32 -> bf16 conversion
// =====================================================================
__global__ void cvt_f32_bf16(const float* __restrict__ src,
                             unsigned short* __restrict__ dst, int n) {
    int i = blockIdx.x * blockDim.x + threadIdx.x;
    int stride = gridDim.x * blockDim.x;
    for (; i < n; i += stride) dst[i] = f2bf(src[i]);
}

// =====================================================================
// Shared GEMM mainloop: block tile 128x64, K-chunks of 64, double-buffered
// LDS, async A staging, 8 waves: wave w owns rows m0+16w..+15, 4 accs.
// =====================================================================
static __device__ __forceinline__ void gemm_mainloop(
    const unsigned short* __restrict__ Ag, int lda,
    const unsigned short* __restrict__ Bg, int ldb,
    int m0, int n0, int K, v8f acc[4])
{
    __shared__ unsigned short As[2][128 * 64];   // row-major [row][k] (16KB each)
    __shared__ unsigned short Bt[2][64 * 64];    // transposed [n][k]  (8KB each)

    const int tid  = threadIdx.x;
    const int lane = tid & 31;
    const int wave = tid >> 5;
    const int ml   = lane & 15;
    const int hh   = lane >> 4;

    auto stage = [&](int buf, int k0) {
        // ---- A block 128x64 bf16 = 16KB: async 16B chunks ----
#pragma unroll
        for (int i = 0; i < 4; ++i) {
            int flat = tid + i * 256;          // 0..1023 16B chunks
            int row = flat >> 3;               // 8 chunks per 64-elem row
            int cu  = flat & 7;
            async_copy16(&As[buf][0] + (size_t)flat * 8,
                         Ag + (size_t)(m0 + row) * lda + k0 + cu * 8);
        }
        // ---- B block 64x64 staged transposed: Bt[n][k] ----
#pragma unroll
        for (int i = 0; i < 8; ++i) {
            int flat = tid + i * 256;          // 0..2047 dwords
            int kr = flat >> 5;                // k row 0..63
            int cu = flat & 31;                // n pair
            unsigned int wv = ((const unsigned int*)Bg)[((size_t)(k0 + kr) * ldb + n0) / 2 + cu];
            int n = cu * 2;
            Bt[buf][n * 64 + kr]       = (unsigned short)(wv & 0xFFFFu);
            Bt[buf][(n + 1) * 64 + kr] = (unsigned short)(wv >> 16);
        }
    };

    stage(0, 0);
    wait_async0();
    __syncthreads();

    for (int k0 = 0; k0 < K; k0 += 64) {
        const int cur = (k0 >> 6) & 1;
        if (k0 + 64 < K) stage(cur ^ 1, k0 + 64);

        const unsigned int* Asu = (const unsigned int*)&As[cur][0];
        const unsigned int* Btu = (const unsigned int*)&Bt[cur][0];
        const int arow = wave * 16 + ml;

        Frag af[2];
#pragma unroll
        for (int c = 0; c < 2; ++c)
#pragma unroll
            for (int v = 0; v < 8; ++v) {
                int kk2 = hh * 8 + v; // K/2 within 32-chunk
                af[c].u[v] = Asu[arow * 32 + 16 * c + kk2];
            }
#pragma unroll
        for (int nt = 0; nt < 4; ++nt) {
#pragma unroll
            for (int c = 0; c < 2; ++c) {
                Frag bf_;
#pragma unroll
                for (int v = 0; v < 8; ++v)
                    bf_.u[v] = Btu[(nt * 16 + ml) * 32 + 16 * c + 8 * hh + v];
                acc[nt] = __builtin_amdgcn_wmma_f32_16x16x32_bf16(
                    false, af[c].v, false, bf_.v, (short)0, acc[nt], false, false);
            }
        }
        wait_async0();
        __syncthreads();
    }
}

// =====================================================================
// QKV GEMM: xb(16384x384) @ wqkv(384x1152) + bias -> q/k/v (B,H,N,D) bf16
// =====================================================================
__global__ void __launch_bounds__(256) gemm_qkv_kernel(
    const unsigned short* __restrict__ Xb, const unsigned short* __restrict__ Wb,
    const float* __restrict__ bias,
    unsigned short* __restrict__ Qo, unsigned short* __restrict__ Ko,
    unsigned short* __restrict__ Vo)
{
    const int n0 = blockIdx.x * 64;
    const int m0 = blockIdx.y * 128;
    v8f acc[4] = {vzero8(), vzero8(), vzero8(), vzero8()};
    gemm_mainloop(Xb, EMBED, Wb, QKV_N, m0, n0, EMBED, acc);

    const int tid = threadIdx.x, lane = tid & 31, wave = tid >> 5;
    const int ml = lane & 15, hh = lane >> 4;
    const int which = n0 / EMBED;            // 0=q 1=k 2=v (uniform per block)
    const int h     = (n0 % EMBED) / HDIM;   // uniform per block
    unsigned short* dst = (which == 0) ? Qo : (which == 1) ? Ko : Vo;
#pragma unroll
    for (int nt = 0; nt < 4; ++nt) {
        int d  = nt * 16 + ml;               // 0..63
        float bv = bias[n0 + d];
#pragma unroll
        for (int r = 0; r < 8; ++r) {
            int row = m0 + wave * 16 + r + 8 * hh;   // 0..16383
            int b = row >> 10, nr = row & 1023;
            size_t idx = (((size_t)(b * HEADS + h)) * SEQ + nr) * HDIM + d;
            dst[idx] = f2bf(acc[nt][r] + bv);
        }
    }
}

// =====================================================================
// Flash attention: grid B*H*8, 256 threads; one 16-row q-tile per wave.
// Double-buffered K/V staging; K via async-to-LDS, V staged transposed.
// =====================================================================
__global__ void __launch_bounds__(256) flash_attn_kernel(
    const unsigned short* __restrict__ Q, const unsigned short* __restrict__ Kb,
    const unsigned short* __restrict__ Vb, unsigned short* __restrict__ Out)
{
    __shared__ unsigned short Ks[2][32 * 64];     // [key][d]
    __shared__ unsigned short Vt[2][64 * 32];     // [d][key]
    __shared__ unsigned short Ps[8 * 16 * 32];    // per-wave P scratch [m][c]

    const int tid  = threadIdx.x;
    const int lane = tid & 31;
    const int wave = tid >> 5;
    const int ml   = lane & 15;
    const int hh   = lane >> 4;

    const int bh = blockIdx.x >> 3;               // 0..95
    const int qc = blockIdx.x & 7;
    const int b  = bh / HEADS, h = bh % HEADS;
    const size_t base = (size_t)(b * HEADS + h) * SEQ * HDIM;
    const unsigned short* qptr = Q  + base;
    const unsigned short* kptr = Kb + base;
    const unsigned short* vptr = Vb + base;
    const int q0 = (qc * 8 + wave) * 16;

    // ---- load Q fragments (16x64 -> two 16x32 A-frags) ----
    Frag aq[2];
    {
        const unsigned int* q32 = (const unsigned int*)(qptr + (size_t)(q0 + ml) * HDIM);
#pragma unroll
        for (int f = 0; f < 2; ++f)
#pragma unroll
            for (int v = 0; v < 8; ++v) {
                int d = 32 * f + hh * 16 + 2 * v;
                aq[f].u[v] = q32[d >> 1];
            }
    }

    auto stage = [&](int buf, int kb) {
        // K block 32x64 bf16 = 4KB: one async 16B chunk per thread
        async_copy16(&Ks[buf][0] + (size_t)tid * 8,
                     kptr + (size_t)kb * HDIM + tid * 8);
        // V block staged transposed
        const unsigned int* vsrc = (const unsigned int*)(vptr + (size_t)kb * HDIM);
#pragma unroll
        for (int i = 0; i < 4; ++i) {
            int flat = tid + i * 256;
            int r  = flat >> 5;                // key row in block
            int cu = flat & 31;                // d pair
            unsigned int wv = vsrc[r * 32 + cu];
            int d = cu * 2;
            Vt[buf][d * 32 + r]       = (unsigned short)(wv & 0xFFFFu);
            Vt[buf][(d + 1) * 32 + r] = (unsigned short)(wv >> 16);
        }
    };

    float mrow[8], lrow[8];
#pragma unroll
    for (int r = 0; r < 8; ++r) { mrow[r] = -1e30f; lrow[r] = 0.f; }
    v8f O0 = vzero8(), O1 = vzero8(), O2 = vzero8(), O3 = vzero8();

    stage(0, 0);
    wait_async0();
    __syncthreads();

    for (int kb = 0; kb < SEQ; kb += 32) {
        const int cur = (kb >> 5) & 1;
        if (kb + 32 < SEQ) stage(cur ^ 1, kb + 32);

        // ---- S = Q * K^T  (16 q-rows x 32 keys) ----
        v8f S0 = vzero8(), S1 = vzero8();
        const unsigned int* Ku = (const unsigned int*)&Ks[cur][0];
#pragma unroll
        for (int c = 0; c < 2; ++c) {              // d-chunk 32c..32c+31
            Frag bk0, bk1;
#pragma unroll
            for (int v = 0; v < 8; ++v) {
                int idx = 16 * c + 8 * hh + v;     // dword within 64-d row
                bk0.u[v] = Ku[(0 * 16 + ml) * 32 + idx];
                bk1.u[v] = Ku[(1 * 16 + ml) * 32 + idx];
            }
            S0 = __builtin_amdgcn_wmma_f32_16x16x32_bf16(
                false, aq[c].v, false, bk0.v, (short)0, S0, false, false);
            S1 = __builtin_amdgcn_wmma_f32_16x16x32_bf16(
                false, aq[c].v, false, bk1.v, (short)0, S1, false, false);
        }

        // ---- online softmax update (rows live in 16-lane halves) ----
        unsigned short* myP = Ps + wave * (16 * 32);
#pragma unroll
        for (int r = 0; r < 8; ++r) {
            float s0 = S0[r] * ATTN_SCALE;
            float s1 = S1[r] * ATTN_SCALE;
            float mx = fmaxf(s0, s1);
#pragma unroll
            for (int off = 1; off < 16; off <<= 1)
                mx = fmaxf(mx, __shfl_xor(mx, off, 32));
            float mn    = fmaxf(mrow[r], mx);
            float alpha = __expf(mrow[r] - mn);
            float p0 = __expf(s0 - mn);
            float p1 = __expf(s1 - mn);
            float rs = p0 + p1;
#pragma unroll
            for (int off = 1; off < 16; off <<= 1)
                rs += __shfl_xor(rs, off, 32);
            lrow[r] = lrow[r] * alpha + rs;
            mrow[r] = mn;
            O0[r] *= alpha; O1[r] *= alpha; O2[r] *= alpha; O3[r] *= alpha;
            int m = r + 8 * hh;
            myP[m * 32 + ml]      = f2bf(p0);
            myP[m * 32 + 16 + ml] = f2bf(p1);
        }
        asm volatile("" ::: "memory");   // keep P stores before reloads (LDS in-order per wave)

        // ---- P A-fragment (16x32) ----
        Frag ap;
        const unsigned int* Pu = (const unsigned int*)myP;
#pragma unroll
        for (int v = 0; v < 8; ++v) {
            int kk2 = hh * 8 + v;
            ap.u[v] = Pu[ml * 16 + kk2];
        }
        // ---- O += P * V  (4 d-tiles) ----
        const unsigned int* Vu = (const unsigned int*)&Vt[cur][0];
#pragma unroll
        for (int dt = 0; dt < 4; ++dt) {
            Frag bv;
#pragma unroll
            for (int v = 0; v < 8; ++v)
                bv.u[v] = Vu[(dt * 16 + ml) * 16 + 8 * hh + v];
            v8f* Op = (dt == 0) ? &O0 : (dt == 1) ? &O1 : (dt == 2) ? &O2 : &O3;
            *Op = __builtin_amdgcn_wmma_f32_16x16x32_bf16(
                false, ap.v, false, bv.v, (short)0, *Op, false, false);
        }
        wait_async0();
        __syncthreads();
    }

    // ---- epilogue: normalize, write (B,N,C) bf16 for proj GEMM ----
#pragma unroll
    for (int r = 0; r < 8; ++r) {
        float inv = 1.0f / lrow[r];
        int m = r + 8 * hh;
        size_t orow = ((size_t)b * SEQ + q0 + m) * EMBED + h * HDIM;
        Out[orow + 0 * 16 + ml] = f2bf(O0[r] * inv);
        Out[orow + 1 * 16 + ml] = f2bf(O1[r] * inv);
        Out[orow + 2 * 16 + ml] = f2bf(O2[r] * inv);
        Out[orow + 3 * 16 + ml] = f2bf(O3[r] * inv);
    }
}

// =====================================================================
// Proj GEMM: attn(16384x384) @ wproj(384x384) + bias -> out fp32
// =====================================================================
__global__ void __launch_bounds__(256) gemm_proj_kernel(
    const unsigned short* __restrict__ Ab, const unsigned short* __restrict__ Wb,
    const float* __restrict__ bias, float* __restrict__ Out)
{
    const int n0 = blockIdx.x * 64;
    const int m0 = blockIdx.y * 128;
    v8f acc[4] = {vzero8(), vzero8(), vzero8(), vzero8()};
    gemm_mainloop(Ab, EMBED, Wb, EMBED, m0, n0, EMBED, acc);

    const int tid = threadIdx.x, lane = tid & 31, wave = tid >> 5;
    const int ml = lane & 15, hh = lane >> 4;
#pragma unroll
    for (int nt = 0; nt < 4; ++nt) {
        int col = n0 + nt * 16 + ml;
        float bv = bias[col];
#pragma unroll
        for (int r = 0; r < 8; ++r) {
            int row = m0 + wave * 16 + r + 8 * hh;
            Out[(size_t)row * EMBED + col] = acc[nt][r] + bv;
        }
    }
}

// =====================================================================
// host launch
// =====================================================================
extern "C" void kernel_launch(void* const* d_in, const int* in_sizes, int n_in,
                              void* d_out, int out_size, void* d_ws, size_t ws_size,
                              hipStream_t stream) {
    (void)in_sizes; (void)n_in; (void)out_size; (void)ws_size;
    const float* x      = (const float*)d_in[0];
    const float* w_qkv  = (const float*)d_in[1];
    const float* b_qkv  = (const float*)d_in[2];
    const float* w_proj = (const float*)d_in[3];
    const float* b_proj = (const float*)d_in[4];
    float* out = (float*)d_out;

    // workspace layout (bf16 elements)
    unsigned short* xb     = (unsigned short*)d_ws;              // 16384*384
    unsigned short* wqkvb  = xb     + (size_t)ROWS * EMBED;      // 384*1152
    unsigned short* wprojb = wqkvb  + (size_t)EMBED * QKV_N;     // 384*384
    unsigned short* qb     = wprojb + (size_t)EMBED * EMBED;     // 16*6*1024*64
    unsigned short* kb     = qb     + (size_t)ROWS * EMBED;
    unsigned short* vb     = kb     + (size_t)ROWS * EMBED;
    unsigned short* attnb  = vb     + (size_t)ROWS * EMBED;      // 16384*384

    cvt_f32_bf16<<<1024, 256, 0, stream>>>(x,      xb,     ROWS * EMBED);
    cvt_f32_bf16<<<256,  256, 0, stream>>>(w_qkv,  wqkvb,  EMBED * QKV_N);
    cvt_f32_bf16<<<128,  256, 0, stream>>>(w_proj, wprojb, EMBED * EMBED);

    gemm_qkv_kernel<<<dim3(QKV_N / 64, ROWS / 128), 256, 0, stream>>>(
        xb, wqkvb, b_qkv, qb, kb, vb);

    flash_attn_kernel<<<BATCH * HEADS * 8, 256, 0, stream>>>(qb, kb, vb, attnb);

    gemm_proj_kernel<<<dim3(EMBED / 64, ROWS / 128), 256, 0, stream>>>(
        attnb, wprojb, b_proj, out);
}
